// SolarRingContextual_90546500535066
// MI455X (gfx1250) — matrix operand order repaired and checked by reading, not compiled, not run on hardware
//
#include <hip/hip_runtime.h>
#include <cstdint>
#include <cstddef>

typedef __attribute__((ext_vector_type(16))) __bf16 v16bf;
typedef __attribute__((ext_vector_type(8)))  __bf16 v8bf;
typedef __attribute__((ext_vector_type(8)))  float  v8f;

#define B_SZ     512
#define L_SZ     256
#define DIN      384
#define DHID     300
#define NSLOT    13
#define NLAY     8

#define NT_H     19    // ceil(300/16) n-tiles for hidden dim
#define KT_EMB   12    // 384/32 k-tiles over embedding dim
#define KT_R     10    // ceil(300/32) k-tiles over hidden dim (padded to 320)
#define KT_TOT   (KT_EMB + KT_R)
#define NT_C     24    // 384/16 n-tiles for output dim
#define KT_MEM   130   // 13 slots * 10 k-tiles
#define KPAD     320

#if __has_builtin(__builtin_amdgcn_tanhf)
#define TANH(x) __builtin_amdgcn_tanhf(x)
#else
#define TANH(x) tanhf(x)
#endif

// ---------------------------------------------------------------- utilities

__global__ void zero_f32(float* __restrict__ p, size_t n) {
  size_t i = (size_t)blockIdx.x * blockDim.x + threadIdx.x;
  size_t s = (size_t)gridDim.x * blockDim.x;
  for (; i < n; i += s) p[i] = 0.f;
}

// Wfused = in_W @ Wh[0];  Wskipf = in_W @ skip_W;
// biasf  = lb[0] + in_b @ Wh[0];  cbias = out_b + skip_b + in_b @ skip_W
__global__ void fuse_weights(const float* __restrict__ inW, const float* __restrict__ Wh0,
                             const float* __restrict__ skipW, const float* __restrict__ inb,
                             const float* __restrict__ lb0, const float* __restrict__ outb,
                             const float* __restrict__ skipb,
                             float* __restrict__ Wfused, float* __restrict__ Wskipf,
                             float* __restrict__ biasf, float* __restrict__ cbias) {
  const int N1 = DIN * DHID, N2 = DIN * DIN;
  int idx = blockIdx.x * blockDim.x + threadIdx.x;
  if (idx < N1) {
    int k = idx / DHID, n = idx % DHID;
    float acc = 0.f;
    for (int d = 0; d < DHID; ++d) acc += inW[k * DHID + d] * Wh0[d * DHID + n];
    Wfused[idx] = acc;
  } else if (idx < N1 + N2) {
    int j = idx - N1; int k = j / DIN, n = j % DIN;
    float acc = 0.f;
    for (int d = 0; d < DHID; ++d) acc += inW[k * DHID + d] * skipW[d * DIN + n];
    Wskipf[j] = acc;
  } else if (idx < N1 + N2 + DHID) {
    int n = idx - N1 - N2;
    float acc = lb0[n];
    for (int d = 0; d < DHID; ++d) acc += inb[d] * Wh0[d * DHID + n];
    biasf[n] = acc;
  } else if (idx < N1 + N2 + DHID + DIN) {
    int n = idx - N1 - N2 - DHID;
    float acc = outb[n] + skipb[n];
    for (int d = 0; d < DHID; ++d) acc += inb[d] * skipW[d * DIN + n];
    cbias[n] = acc;
  }
}

// Pack a KxN (row-major, leading dim ld) f32 matrix into WMMA bf16 B-fragment
// order: frag(kt,nt) = 32 lanes x 16 bf16 contiguous; lane L: n = nt*16+L%16,
// k = kt*32 + 8*(L/16) + (j&7) + 16*(j/8)  (j=0..15, matching VGPR pair order).
__global__ void pack_b_linear(const float* __restrict__ src, __bf16* __restrict__ dst,
                              int Krows, int Ncols, int ld, int KT, int NT) {
  int total = KT * NT * 32 * 16;
  for (int idx = blockIdx.x * blockDim.x + threadIdx.x; idx < total;
       idx += gridDim.x * blockDim.x) {
    int j    = idx & 15;
    int lane = (idx >> 4) & 31;
    int nt   = (idx >> 9) % NT;
    int kt   = (idx >> 9) / NT;
    int n = nt * 16 + (lane & 15);
    int k = kt * 32 + 8 * (lane >> 4) + (j & 7) + 16 * (j >> 3);
    float v = (k < Krows && n < Ncols) ? src[(size_t)k * ld + n] : 0.f;
    dst[idx] = (__bf16)v;
  }
}

// out_W rows used are only layer-0 rows: flat row = (s*8+0)*300 + d.
// Packed K axis = 13 slots x 320 (d padded 300->320).
__global__ void pack_b_out(const float* __restrict__ src, __bf16* __restrict__ dst) {
  const int KT = KT_MEM, NT = NT_C;
  int total = KT * NT * 32 * 16;
  for (int idx = blockIdx.x * blockDim.x + threadIdx.x; idx < total;
       idx += gridDim.x * blockDim.x) {
    int j    = idx & 15;
    int lane = (idx >> 4) & 31;
    int nt   = (idx >> 9) % NT;
    int kt   = (idx >> 9) / NT;
    int n  = nt * 16 + (lane & 15);
    int kg = kt * 32 + 8 * (lane >> 4) + (j & 7) + 16 * (j >> 3);
    int s = kg / KPAD, d = kg % KPAD;
    float v = 0.f;
    if (d < DHID) v = src[(size_t)(s * (NLAY * DHID) + d) * DIN + n];
    dst[idx] = (__bf16)v;
  }
}

// ---------------------------------------------------------------- fragment loads

__device__ __forceinline__ v16bf load_b_frag(const __bf16* __restrict__ base,
                                             int kt, int nt, int NT, int lane) {
  const __bf16* p = base + ((((size_t)kt * NT + nt) * 32 + lane) << 4);
  v8bf lo = *(const v8bf*)p;
  v8bf hi = *(const v8bf*)(p + 8);
  v16bf r;
#pragma unroll
  for (int j = 0; j < 8; ++j) { r[j] = lo[j]; r[j + 8] = hi[j]; }
  return r;
}

// A fragment, 16-bit layout: element j<8 -> k = kbase + 8*half + j,
// j>=8 -> k = kbase + 16 + 8*half + (j-8). Source: bf16 row in LDS.
__device__ __forceinline__ v16bf load_a_lds(const __bf16* row, int kbase, int half) {
  v8bf lo = *(const v8bf*)(row + kbase + 8 * half);
  v8bf hi = *(const v8bf*)(row + kbase + 16 + 8 * half);
  v16bf r;
#pragma unroll
  for (int j = 0; j < 8; ++j) { r[j] = lo[j]; r[j + 8] = hi[j]; }
  return r;
}

// Same A layout but converting an f32 global row on the fly.
__device__ __forceinline__ v16bf load_a_f32(const float* __restrict__ row,
                                            int kbase, int half) {
  const float* p0 = row + kbase + 8 * half;
  const float* p1 = row + kbase + 16 + 8 * half;
  v16bf r;
#pragma unroll
  for (int j = 0; j < 8; ++j) { r[j] = (__bf16)p0[j]; r[j + 8] = (__bf16)p1[j]; }
  return r;
}

#define WMMA_BF16(A, Bf, C) \
  __builtin_amdgcn_wmma_f32_16x16x32_bf16(false, (A), false, (Bf), (short)0, (C), false, false)

// ---------------------------------------------------------------- recurrence
// One WG owns 16 batch rows. Per step: h0 = tanh(emb_t @ Wfused + (S/13) @ Wr0 + biasf)
// S (f32, LDS) is the running sum of the 13 ring slots; slots live in d_out's mem region.
// The 22 k-tiles per step are software-pipelined: fragments for kt+1 are issued
// before the WMMAs of kt, so the s_wait_loadcnt retires behind matrix work.
__global__ void __launch_bounds__(256)
recurrence_kernel(const float* __restrict__ emb,
                  const __bf16* __restrict__ pWf,
                  const __bf16* __restrict__ pWr,
                  const float* __restrict__ biasf,
                  float* __restrict__ mem_out) {
  __shared__ __align__(16) float  S[16][304];
  __shared__ __align__(16) __bf16 rbuf[16][KPAD];

  const int tid = threadIdx.x;
  const int wave = tid >> 5, lane = tid & 31;
  const int half = lane >> 4, lq = lane & 15;
  const int wg = blockIdx.x;

  const int nt0 = wave, nt1 = wave + 8, nt2 = wave + 16;
  const bool have3 = (nt2 < NT_H);

  for (int i = tid; i < 16 * 304; i += 256) (&S[0][0])[i] = 0.f;
  for (int i = tid; i < 16 * KPAD; i += 256) (&rbuf[0][0])[i] = (__bf16)0.f;
  __syncthreads();

  const int n0 = nt0 * 16 + lq, n1 = nt1 * 16 + lq, n2 = nt2 * 16 + lq;
  const float bs0 = (n0 < DHID) ? biasf[n0] : 0.f;
  const float bs1 = (n1 < DHID) ? biasf[n1] : 0.f;
  const float bs2 = (have3 && n2 < DHID) ? biasf[n2] : 0.f;

  const float* embrow = emb + (size_t)(wg * 16 + lq) * L_SZ * DIN;

  for (int t = 0; t < L_SZ; ++t) {
    const int ptr = t % NSLOT;
    const float* xrow = embrow + (size_t)t * DIN;
    if (t + 1 < L_SZ) {                       // hide next step's A stream
      __builtin_prefetch(xrow + DIN, 0, 1);
      __builtin_prefetch(xrow + DIN + 192, 0, 1);
    }

    v8f c0 = {0.f, 0.f, 0.f, 0.f, 0.f, 0.f, 0.f, 0.f};
    v8f c1 = c0, c2 = c0;

    auto load_a_any = [&](int kt) -> v16bf {
      return (kt < KT_EMB) ? load_a_f32(xrow, kt * 32, half)
                           : load_a_lds(&rbuf[lq][0], (kt - KT_EMB) * 32, half);
    };
    auto load_b3 = [&](int kt, v16bf& b0, v16bf& b1, v16bf& b2) {
      const __bf16* base = (kt < KT_EMB) ? pWf : pWr;
      int k = (kt < KT_EMB) ? kt : kt - KT_EMB;
      b0 = load_b_frag(base, k, nt0, NT_H, lane);
      b1 = load_b_frag(base, k, nt1, NT_H, lane);
      if (have3) b2 = load_b_frag(base, k, nt2, NT_H, lane);
    };

    // prologue
    v16bf a_cur = load_a_any(0);
    v16bf b0c = {}, b1c = {}, b2c = {};
    load_b3(0, b0c, b1c, b2c);

    for (int kt = 0; kt < KT_TOT; ++kt) {
      v16bf a_nx = {}, b0n = {}, b1n = {}, b2n = {};
      if (kt + 1 < KT_TOT) {                  // issue next tile's loads first
        a_nx = load_a_any(kt + 1);
        load_b3(kt + 1, b0n, b1n, b2n);
      }
      c0 = WMMA_BF16(a_cur, b0c, c0);
      c1 = WMMA_BF16(a_cur, b1c, c1);
      if (have3) c2 = WMMA_BF16(a_cur, b2c, c2);
      a_cur = a_nx; b0c = b0n; b1c = b1n; b2c = b2n;
    }
    __syncthreads();  // all rbuf reads complete before ring update

    auto epilogue = [&](v8f& c, int n, float bs) {
      if (n >= DHID) return;
#pragma unroll
      for (int r = 0; r < 8; ++r) {
        int m = r + 8 * half;
        float h = TANH(c[r] + bs);
        float* slot = mem_out +
            ((size_t)(wg * 16 + m) * NSLOT + ptr) * (NLAY * DHID) + n;  // layer 0
        float old = (t >= NSLOT) ? *slot : 0.f;
        *slot = h;
        float s = S[m][n] + h - old;
        S[m][n] = s;
        rbuf[m][n] = (__bf16)(s * (1.0f / 13.0f));
      }
    };
    epilogue(c0, n0, bs0);
    epilogue(c1, n1, bs1);
    if (have3) epilogue(c2, n2, bs2);
    __syncthreads();
  }
}

// ---------------------------------------------------------------- finalizer
// c = LN( mem_slots @ outW_layer0 + emb0 @ Wskipf + cbias ) ; logit = MLP head
__global__ void __launch_bounds__(256)
final_kernel(const float* __restrict__ emb,
             const float* __restrict__ mem,
             const __bf16* __restrict__ pWout,
             const __bf16* __restrict__ pWskip,
             const float* __restrict__ cbias,
             const float* __restrict__ ln_g, const float* __restrict__ ln_b,
             const float* __restrict__ shW1, const float* __restrict__ shb1,
             const float* __restrict__ shW2, const float* __restrict__ shb2,
             float* __restrict__ out_c, float* __restrict__ out_logit) {
  __shared__ __align__(16) float cbuf[16][DIN];
  const int tid = threadIdx.x;
  const int wave = tid >> 5, lane = tid & 31;
  const int half = lane >> 4, lq = lane & 15;
  const int wg = blockIdx.x;

  const int nt0 = wave, nt1 = wave + 8, nt2 = wave + 16;  // NT_C = 24, always 3
  v8f c0 = {0.f, 0.f, 0.f, 0.f, 0.f, 0.f, 0.f, 0.f};
  v8f c1 = c0, c2 = c0;
  const size_t brow = (size_t)(wg * 16 + lq);

  // 13 ring slots (layer 0 of mem) against packed out_W rows
  for (int kt = 0; kt < KT_MEM; ++kt) {
    int s = kt / KT_R, dbase = (kt % KT_R) * 32;
    const float* mrow = mem + (brow * NSLOT + s) * (NLAY * DHID);  // layer 0
    v16bf a;
#pragma unroll
    for (int j = 0; j < 8; ++j) {
      int d0 = dbase + 8 * half + j;
      int d1 = dbase + 16 + 8 * half + j;
      a[j]     = (__bf16)((d0 < DHID) ? mrow[d0] : 0.f);
      a[j + 8] = (__bf16)((d1 < DHID) ? mrow[d1] : 0.f);
    }
    v16bf b = load_b_frag(pWout, kt, nt0, NT_C, lane);
    c0 = WMMA_BF16(a, b, c0);
    b = load_b_frag(pWout, kt, nt1, NT_C, lane);
    c1 = WMMA_BF16(a, b, c1);
    b = load_b_frag(pWout, kt, nt2, NT_C, lane);
    c2 = WMMA_BF16(a, b, c2);
  }
  // skip path: emb0 @ Wskipf
  const float* erow = emb + brow * L_SZ * DIN;
  for (int kt = 0; kt < KT_EMB; ++kt) {
    v16bf a = load_a_f32(erow, kt * 32, half);
    v16bf b = load_b_frag(pWskip, kt, nt0, NT_C, lane);
    c0 = WMMA_BF16(a, b, c0);
    b = load_b_frag(pWskip, kt, nt1, NT_C, lane);
    c1 = WMMA_BF16(a, b, c1);
    b = load_b_frag(pWskip, kt, nt2, NT_C, lane);
    c2 = WMMA_BF16(a, b, c2);
  }

  auto spill = [&](v8f& c, int nt) {
    int n = nt * 16 + lq;
    float cb = cbias[n];
#pragma unroll
    for (int r = 0; r < 8; ++r) cbuf[r + 8 * half][n] = c[r] + cb;
  };
  spill(c0, nt0); spill(c1, nt1); spill(c2, nt2);
  __syncthreads();

  if (tid < 16) {
    const int m = tid, b = wg * 16 + m;
    float mu = 0.f;
    for (int j = 0; j < DIN; ++j) mu += cbuf[m][j];
    mu *= (1.0f / DIN);
    float var = 0.f;
    for (int j = 0; j < DIN; ++j) { float d = cbuf[m][j] - mu; var += d * d; }
    var *= (1.0f / DIN);
    float inv = rsqrtf(var + 1e-5f);
    for (int j = 0; j < DIN; ++j) {
      float cv = (cbuf[m][j] - mu) * inv * ln_g[j] + ln_b[j];
      cbuf[m][j] = cv;
      out_c[(size_t)b * DIN + j] = cv;
    }
    float lg = 0.f;
    for (int u = 0; u < 64; ++u) {
      float a2 = shb1[u];
      for (int j = 0; j < DIN; ++j) a2 += cbuf[m][j] * shW1[j * 64 + u];
      lg += (a2 > 0.f ? a2 : 0.f) * shW2[u];
    }
    out_logit[b] = lg + shb2[0];
  }
}

// ---------------------------------------------------------------- launch

extern "C" void kernel_launch(void* const* d_in, const int* in_sizes, int n_in,
                              void* d_out, int out_size, void* d_ws, size_t ws_size,
                              hipStream_t stream) {
  (void)in_sizes; (void)n_in; (void)out_size; (void)ws_size;
  const float* emb   = (const float*)d_in[0];
  const float* inW   = (const float*)d_in[1];
  const float* inb   = (const float*)d_in[2];
  const float* Wh    = (const float*)d_in[3];
  const float* Wr    = (const float*)d_in[4];
  const float* lb    = (const float*)d_in[5];
  const float* skipW = (const float*)d_in[6];
  const float* skipb = (const float*)d_in[7];
  const float* outW  = (const float*)d_in[8];
  const float* outb  = (const float*)d_in[9];
  const float* lng   = (const float*)d_in[10];
  const float* lnb   = (const float*)d_in[11];
  const float* shW1  = (const float*)d_in[12];
  const float* shb1  = (const float*)d_in[13];
  const float* shW2  = (const float*)d_in[14];
  const float* shb2  = (const float*)d_in[15];

  float* out_c     = (float*)d_out;                               // (512,384)
  float* out_mem   = out_c + (size_t)B_SZ * DIN;                  // (512,13,8,300)
  float* out_logit = out_mem + (size_t)B_SZ * NSLOT * NLAY * DHID;// (512,1)

  uint8_t* w = (uint8_t*)d_ws;
  auto carve = [&](size_t bytes) -> void* {
    void* p = (void*)w;
    w += (bytes + 255) & ~(size_t)255;
    return p;
  };
  float*  Wfused = (float*)carve((size_t)DIN * DHID * 4);
  float*  Wskipf = (float*)carve((size_t)DIN * DIN * 4);
  float*  biasf  = (float*)carve(304 * 4);
  float*  cbias  = (float*)carve(DIN * 4);
  __bf16* pWf    = (__bf16*)carve((size_t)KT_EMB * NT_H * 512 * 2);
  __bf16* pWr    = (__bf16*)carve((size_t)KT_R * NT_H * 512 * 2);
  __bf16* pWout  = (__bf16*)carve((size_t)KT_MEM * NT_C * 512 * 2);
  __bf16* pWskip = (__bf16*)carve((size_t)KT_EMB * NT_C * 512 * 2);

  zero_f32<<<2048, 256, 0, stream>>>(out_mem, (size_t)B_SZ * NSLOT * NLAY * DHID);

  int fuse_total = DIN * DHID + DIN * DIN + DHID + DIN;
  fuse_weights<<<(fuse_total + 255) / 256, 256, 0, stream>>>(
      inW, Wh, skipW, inb, lb, outb, skipb, Wfused, Wskipf, biasf, cbias);

  pack_b_linear<<<(KT_EMB * NT_H * 512 + 255) / 256, 256, 0, stream>>>(
      Wfused, pWf, DIN, DHID, DHID, KT_EMB, NT_H);
  pack_b_linear<<<(KT_R * NT_H * 512 + 255) / 256, 256, 0, stream>>>(
      Wr, pWr, DHID, DHID, DHID, KT_R, NT_H);   // Wr[0] = first 300x300 block
  pack_b_out<<<4096, 256, 0, stream>>>(outW, pWout);
  pack_b_linear<<<(KT_EMB * NT_C * 512 + 255) / 256, 256, 0, stream>>>(
      Wskipf, pWskip, DIN, DIN, DIN, KT_EMB, NT_C);

  recurrence_kernel<<<32, 256, 0, stream>>>(emb, pWf, pWr, biasf, out_mem);
  final_kernel<<<32, 256, 0, stream>>>(emb, out_mem, pWout, pWskip, cbias,
                                       lng, lnb, shW1, shb1, shW2, shb2,
                                       out_c, out_logit);
}
